// _PointnetSAModuleBase1_29540785061889
// MI455X (gfx1250) — matrix operand order repaired and checked by reading, not compiled, not run on hardware
//
#include <hip/hip_runtime.h>

#define B_ 16
#define N_ 4096
#define C_ 64
#define NPOINT_ 1024
#define NSAMPLE_ 32
#define R2_ 0.04f
#define TS_ 4   // centroids per MLP workgroup (128 rows = 8 M-tiles)

typedef __attribute__((ext_vector_type(16))) _Float16 v16h;
typedef __attribute__((ext_vector_type(8)))  _Float16 v8h;
typedef __attribute__((ext_vector_type(8)))  float    v8f;

// -------- WMMA fragment loaders (16x16x32 f16) --------
// A (16x32): lane L -> row M=L%16; lanes<16 hold K {0..7, 16..23}, lanes>=16 hold K {8..15, 24..31}
__device__ __forceinline__ v16h frag_a(const _Float16* row0, int ld, int lane, int k0) {
  const _Float16* p = row0 + (lane & 15) * ld + k0 + ((lane & 16) ? 8 : 0);
  v8h lo = *(const v8h*)(p);
  v8h hi = *(const v8h*)(p + 16);
  v16h r;
#pragma unroll
  for (int i = 0; i < 8; ++i) { r[i] = lo[i]; r[i + 8] = hi[i]; }
  return r;
}

// B (32x16): lane L -> col N=L%16; lanes<16 hold K {0..7, 16..23}, lanes>=16 hold K {8..15, 24..31}
__device__ __forceinline__ v16h frag_b(const _Float16* w, int ldn, int lane, int k0, int n0) {
  const _Float16* p = w + (size_t)(k0 + ((lane & 16) ? 8 : 0)) * ldn + n0 + (lane & 15);
  v16h r;
#pragma unroll
  for (int i = 0; i < 8; ++i) { r[i] = p[(size_t)i * ldn]; r[i + 8] = p[(size_t)(16 + i) * ldn]; }
  return r;
}

// -------- Kernel 0: convert weights to f16, pad W1's K 67->96 --------
__global__ __launch_bounds__(256) void prep_weights(
    const float* __restrict__ W1, const float* __restrict__ W2, const float* __restrict__ W3,
    _Float16* __restrict__ w1h, _Float16* __restrict__ w2h, _Float16* __restrict__ w3h) {
  int tid = threadIdx.x;
  for (int i = tid; i < 96 * 64; i += 256) {
    int k = i >> 6;
    w1h[i] = (k < 67) ? (_Float16)W1[i] : (_Float16)0.0f;
  }
  for (int i = tid; i < 64 * 64; i += 256) w2h[i] = (_Float16)W2[i];
  for (int i = tid; i < 64 * 128; i += 256) w3h[i] = (_Float16)W3[i];
}

// -------- Kernel 1: furthest point sampling (4-D pts: xyz + masked curvature) --------
__global__ __launch_bounds__(256) void fps_kernel(
    const float* __restrict__ xyz, const float* __restrict__ curv,
    float* __restrict__ out_xyz, float* __restrict__ out_curv) {
  int b = blockIdx.x, tid = threadIdx.x;
  float px[16], py[16], pz[16], pc[16], md[16];
#pragma unroll
  for (int i = 0; i < 16; ++i) {
    int n = i * 256 + tid;
    const float* p = xyz + ((size_t)b * N_ + n) * 3;
    float x = p[0], y = p[1], z = p[2];
    float c = curv[(size_t)b * N_ + n];
    if (x * x + y * y + z * z > 0.49f) c = 0.0f;   // ||xyz|| > 0.7
    px[i] = x; py[i] = y; pz[i] = z; pc[i] = c; md[i] = 1e10f;
  }
  __shared__ float bc[4];
  __shared__ float rv[8];
  __shared__ int   ri[8];
  __shared__ int   snext;
  int last = 0;
  for (int s = 0; s < NPOINT_; ++s) {
    if (tid == (last & 255)) {
      int slot = last >> 8;
      bc[0] = px[slot]; bc[1] = py[slot]; bc[2] = pz[slot]; bc[3] = pc[slot];
      float* o = out_xyz + ((size_t)b * NPOINT_ + s) * 3;
      o[0] = px[slot]; o[1] = py[slot]; o[2] = pz[slot];
      out_curv[(size_t)b * NPOINT_ + s] = pc[slot];
    }
    __syncthreads();
    float cx = bc[0], cy = bc[1], cz = bc[2], cc = bc[3];
    float bv = -1.0f; int bi = 0x7fffffff;
#pragma unroll
    for (int i = 0; i < 16; ++i) {
      float dx = px[i] - cx, dy = py[i] - cy, dz = pz[i] - cz, dc = pc[i] - cc;
      float d = dx * dx + dy * dy + dz * dz + dc * dc;
      md[i] = fminf(md[i], d);
      if (md[i] > bv) { bv = md[i]; bi = i * 256 + tid; }
    }
#pragma unroll
    for (int off = 16; off > 0; off >>= 1) {
      float ov = __shfl_xor(bv, off, 32);
      int   oi = __shfl_xor(bi, off, 32);
      if (ov > bv || (ov == bv && oi < bi)) { bv = ov; bi = oi; }
    }
    if ((tid & 31) == 0) { rv[tid >> 5] = bv; ri[tid >> 5] = bi; }
    __syncthreads();
    if (tid == 0) {
      float v = rv[0]; int ix = ri[0];
      for (int w = 1; w < 8; ++w)
        if (rv[w] > v || (rv[w] == v && ri[w] < ix)) { v = rv[w]; ix = ri[w]; }
      snext = ix;
    }
    __syncthreads();
    last = snext;
  }
}

// -------- Kernel 2: ball query (first NSAMPLE in-radius indices, ascending) --------
__global__ __launch_bounds__(1024) void ballq_kernel(
    const float* __restrict__ xyz, const float* __restrict__ new_xyz,
    int* __restrict__ gidx) {
  int b = blockIdx.x, tid = threadIdx.x;
  __shared__ float sx[N_], sy[N_], sz[N_];
  for (int n = tid; n < N_; n += 1024) {
    const float* p = xyz + ((size_t)b * N_ + n) * 3;
    sx[n] = p[0]; sy[n] = p[1]; sz[n] = p[2];
  }
  __syncthreads();
  int s = tid;
  const float* c = new_xyz + ((size_t)b * NPOINT_ + s) * 3;
  float cx = c[0], cy = c[1], cz = c[2];
  int* g = gidx + ((size_t)b * NPOINT_ + s) * NSAMPLE_;
  int cnt = 0, first = N_ - 1;
  for (int j = 0; j < N_; ++j) {
    float dx = sx[j] - cx, dy = sy[j] - cy, dz = sz[j] - cz;
    if (dx * dx + dy * dy + dz * dz <= R2_) {
      if (cnt == 0) first = j;
      g[cnt] = j;
      if (++cnt == NSAMPLE_) break;
    }
  }
  for (int k = cnt; k < NSAMPLE_; ++k) g[k] = first;
}

// -------- Kernel 3: fused gather + 3-layer MLP (WMMA) + max-pool + transpose --------
__global__ __launch_bounds__(256) void mlp_kernel(
    const float* __restrict__ xyz, const float* __restrict__ feat,
    const float* __restrict__ b1, const float* __restrict__ b2, const float* __restrict__ b3,
    const _Float16* __restrict__ w1h, const _Float16* __restrict__ w2h, const _Float16* __restrict__ w3h,
    const int* __restrict__ gidx, const float* __restrict__ new_xyz,
    float* __restrict__ out_feat) {
  int stile = blockIdx.x;           // 0..255 (NPOINT_/TS_)
  int b = blockIdx.y;               // 0..15
  int tid = threadIdx.x;
  int lane = tid & 31, w = tid >> 5;

  __shared__ __align__(16) _Float16 sh_in[128 * 96];   // layer-1 A (K padded to 96); reused as layer-2 out [128][64]
  __shared__ __align__(16) _Float16 sh_mid[128 * 64];  // layer-1 out / layer-2 A
  __shared__ int sh_pool[TS_ * 128];

  for (int i = tid; i < TS_ * 128; i += 256) sh_pool[i] = 0;   // 0 bits == 0.0f, valid lower bound after ReLU

  // ---- gather: 128 rows x (3 rel-xyz + 64 feat + 29 zero-pad), 2 threads per row ----
  {
    int r = tid >> 1, half = tid & 1;
    int ct = r >> 5, smp = r & 31;
    int s = stile * TS_ + ct;
    int gi = gidx[((size_t)b * NPOINT_ + s) * NSAMPLE_ + smp];
    const float* cx = new_xyz + ((size_t)b * NPOINT_ + s) * 3;
    const float* px = xyz + ((size_t)b * N_ + gi) * 3;
    const float* pf = feat + ((size_t)b * N_ + gi) * C_;
    int c0 = half * 48;
    for (int c = c0; c < c0 + 48; ++c) {
      float v;
      if (c < 3)       v = px[c] - cx[c];
      else if (c < 67) v = pf[c - 3];
      else             v = 0.0f;
      sh_in[r * 96 + c] = (_Float16)v;
    }
  }
  __syncthreads();

  int m0 = w * 16;               // each wave owns one 16-row M-tile
  int col = lane & 15;
  int hi8 = (lane & 16) ? 8 : 0;

  // ---- Layer 1: [128x96] @ [96x64] ----
  v8f acc1[4];
#pragma unroll
  for (int nt = 0; nt < 4; ++nt) {
    float bv = b1[nt * 16 + col];
#pragma unroll
    for (int r = 0; r < 8; ++r) acc1[nt][r] = bv;
  }
#pragma unroll
  for (int k = 0; k < 3; ++k) {
    v16h a = frag_a(sh_in + m0 * 96, 96, lane, k * 32);
#pragma unroll
    for (int nt = 0; nt < 4; ++nt) {
      v16h bb = frag_b(w1h, 64, lane, k * 32, nt * 16);
      acc1[nt] = __builtin_amdgcn_wmma_f32_16x16x32_f16(false, a, false, bb, (short)0, acc1[nt], false, false);
    }
  }
#pragma unroll
  for (int nt = 0; nt < 4; ++nt)
#pragma unroll
    for (int r = 0; r < 8; ++r) {
      float v = acc1[nt][r]; v = v > 0.0f ? v : 0.0f;
      sh_mid[(m0 + hi8 + r) * 64 + nt * 16 + col] = (_Float16)v;
    }
  __syncthreads();

  // ---- Layer 2: [128x64] @ [64x64], output into sh_in reinterpreted [128][64] ----
  v8f acc2[4];
#pragma unroll
  for (int nt = 0; nt < 4; ++nt) {
    float bv = b2[nt * 16 + col];
#pragma unroll
    for (int r = 0; r < 8; ++r) acc2[nt][r] = bv;
  }
#pragma unroll
  for (int k = 0; k < 2; ++k) {
    v16h a = frag_a(sh_mid + m0 * 64, 64, lane, k * 32);
#pragma unroll
    for (int nt = 0; nt < 4; ++nt) {
      v16h bb = frag_b(w2h, 64, lane, k * 32, nt * 16);
      acc2[nt] = __builtin_amdgcn_wmma_f32_16x16x32_f16(false, a, false, bb, (short)0, acc2[nt], false, false);
    }
  }
#pragma unroll
  for (int nt = 0; nt < 4; ++nt)
#pragma unroll
    for (int r = 0; r < 8; ++r) {
      float v = acc2[nt][r]; v = v > 0.0f ? v : 0.0f;
      sh_in[(m0 + hi8 + r) * 64 + nt * 16 + col] = (_Float16)v;
    }
  __syncthreads();

  // ---- Layer 3: [128x64] @ [64x128], ReLU + max-pool over 32 samples ----
  v8f acc3[8];
#pragma unroll
  for (int nt = 0; nt < 8; ++nt) {
    float bv = b3[nt * 16 + col];
#pragma unroll
    for (int r = 0; r < 8; ++r) acc3[nt][r] = bv;
  }
#pragma unroll
  for (int k = 0; k < 2; ++k) {
    v16h a = frag_a(sh_in + m0 * 64, 64, lane, k * 32);
#pragma unroll
    for (int nt = 0; nt < 8; ++nt) {
      v16h bb = frag_b(w3h, 128, lane, k * 32, nt * 16);
      acc3[nt] = __builtin_amdgcn_wmma_f32_16x16x32_f16(false, a, false, bb, (short)0, acc3[nt], false, false);
    }
  }
#pragma unroll
  for (int nt = 0; nt < 8; ++nt)
#pragma unroll
    for (int r = 0; r < 8; ++r) {
      int row = m0 + hi8 + r;
      int ct = row >> 5;
      float v = acc3[nt][r]; v = v > 0.0f ? v : 0.0f;
      atomicMax(&sh_pool[ct * 128 + nt * 16 + col], __float_as_int(v));
    }
  __syncthreads();

  // ---- write (B, 128, NPOINT) transposed output ----
  for (int i = tid; i < TS_ * 128; i += 256) {
    int ct = i >> 7, ch = i & 127;
    out_feat[(size_t)b * 128 * NPOINT_ + (size_t)ch * NPOINT_ + (stile * TS_ + ct)] =
        __int_as_float(sh_pool[i]);
  }
}

extern "C" void kernel_launch(void* const* d_in, const int* in_sizes, int n_in,
                              void* d_out, int out_size, void* d_ws, size_t ws_size,
                              hipStream_t stream) {
  const float* xyz  = (const float*)d_in[0];
  const float* feat = (const float*)d_in[1];
  const float* curv = (const float*)d_in[2];
  const float* W1   = (const float*)d_in[3];
  const float* b1   = (const float*)d_in[4];
  const float* W2   = (const float*)d_in[5];
  const float* b2   = (const float*)d_in[6];
  const float* W3   = (const float*)d_in[7];
  const float* b3   = (const float*)d_in[8];

  float* out      = (float*)d_out;
  float* out_xyz  = out;                                  // (16,1024,3)
  float* out_feat = out + (size_t)B_ * NPOINT_ * 3;       // (16,128,1024)
  float* out_curv = out_feat + (size_t)B_ * 128 * NPOINT_;// (16,1024)

  char* ws = (char*)d_ws;
  _Float16* w1h = (_Float16*)ws;            // 96*64 f16
  _Float16* w2h = w1h + 96 * 64;            // 64*64 f16
  _Float16* w3h = w2h + 64 * 64;            // 64*128 f16
  int* gidx = (int*)(ws + 36864);           // (16,1024,32) int, 256B aligned

  prep_weights<<<1, 256, 0, stream>>>(W1, W2, W3, w1h, w2h, w3h);
  fps_kernel<<<B_, 256, 0, stream>>>(xyz, curv, out_xyz, out_curv);
  ballq_kernel<<<B_, 1024, 0, stream>>>(xyz, out_xyz, gidx);
  dim3 grid(NPOINT_ / TS_, B_);
  mlp_kernel<<<grid, 256, 0, stream>>>(xyz, feat, b1, b2, b3, w1h, w2h, w3h,
                                       gidx, out_xyz, out_feat);
}
